// SpectraResidualVectorQuantize_13967233647408
// MI455X (gfx1250) — compile-verified, matrix operand order checked by reading, not run on hardware
//
#include <hip/hip_runtime.h>
#include <math.h>

typedef __attribute__((ext_vector_type(2)))  float        v2f;
typedef __attribute__((ext_vector_type(8)))  float        v8f;
typedef __attribute__((ext_vector_type(16))) _Float16     v16h;
typedef __attribute__((ext_vector_type(4)))  unsigned int v4u;
typedef __attribute__((ext_vector_type(8)))  int          v8i_;
typedef __attribute__((ext_vector_type(4)))  int          v4i_;

#define B_  8
#define D_  512
#define T_  4096
#define CD_ 64
#define CS_ 1024
#define NS_ 4

// ---------------------------------------------------------------- utilities
__device__ __forceinline__ float warp_sum(float v) {
  #pragma unroll
  for (int off = 16; off; off >>= 1) v += __shfl_xor(v, off, 32);
  return v;
}

// ---------------------------------------------------------------- TDM (Tensor Data Mover)
// 2D tile load Global->LDS via tensor_load_to_lds (6-arg clang-23 form).
// Descriptor per CDNA5 ISA D# layout:
//  group0: [1:0]=count, [63:32]=lds_addr, [120:64]=global_addr, [127:126]=type(2="image")
//  group1: [17:16]=data_size(2=4B), [20]=pad_en, [24:22]=pad_interval(4 -> 32 DWORDs),
//          [31:25]=pad_amount(0 -> 1 DWORD), [79:48]=tensor_dim0, [111:80]=tensor_dim1,
//          [127:112]=tile_dim0, [143:128]=tile_dim1, [207:160]=tensor_dim0_stride
//  groups 2/3 + trailing group: zero (2D tensor, not in a cluster)
__device__ __forceinline__ void tdm_load_2d(unsigned lds_off, const void* gptr,
                                            unsigned tile_d0, unsigned tile_d1,
                                            unsigned tensor_d0, unsigned tensor_d1,
                                            unsigned long long stride0,
                                            unsigned pad_en, unsigned pad_interval) {
  const unsigned long long ga = (unsigned long long)gptr;
  v4u g0;
  g0[0] = 1u;                                            // count = 1 valid descriptor
  g0[1] = lds_off;                                       // lds_addr (bytes)
  g0[2] = (unsigned)(ga & 0xFFFFFFFFull);                // global_addr[31:0]
  g0[3] = (unsigned)((ga >> 32) & 0x01FFFFFFull)         // global_addr[56:32]
        | (2u << 30);                                    // type = 2
  v8i_ g1;
  g1[0] = (int)((2u << 16) | (pad_en << 20) | (pad_interval << 22));   // 4B elems, padding
  g1[1] = (int)((tensor_d0 & 0xFFFFu) << 16);                          // tensor_dim0 lo16
  g1[2] = (int)((tensor_d0 >> 16) | ((tensor_d1 & 0xFFFFu) << 16));    // td0 hi16 | td1 lo16
  g1[3] = (int)((tensor_d1 >> 16) | (tile_d0 << 16));                  // td1 hi16 | tile_dim0
  g1[4] = (int)(tile_d1 & 0xFFFFu);                                    // tile_dim1 (tile_dim2=0)
  g1[5] = (int)(unsigned)(stride0 & 0xFFFFFFFFull);                    // dim0 stride lo32
  g1[6] = (int)(unsigned)((stride0 >> 32) & 0xFFFFull);                // dim0 stride hi16
  g1[7] = 0;
  const v4i_ gz4 = {0, 0, 0, 0};
  const v8i_ gz8 = {0, 0, 0, 0, 0, 0, 0, 0};
  __builtin_amdgcn_tensor_load_to_lds(g0, g1, gz4, gz4, gz8, 0);
}

// ---------------------------------------------------------------- weight prep
// weight_norm: W[row] = g[row] * v[row] / max(||v[row]||, 1e-12). One wave per row.
__global__ void prep_in_weights(const float* __restrict__ v_in, const float* __restrict__ g_in,
                                float* __restrict__ Wi) {
  const int row = blockIdx.x;                    // NS_*CD_ rows, each length D_
  const float* src = v_in + (size_t)row * D_;
  float s = 0.f;
  for (int d = threadIdx.x; d < D_; d += 32) { float x = src[d]; s += x * x; }
  s = warp_sum(s);
  float nrm = sqrtf(s); if (nrm < 1e-12f) nrm = 1e-12f;
  const float sc = g_in[row] / nrm;
  float* dst = Wi + (size_t)row * D_;
  for (int d = threadIdx.x; d < D_; d += 32) dst[d] = src[d] * sc;
}

__global__ void prep_out_weights(const float* __restrict__ v_out, const float* __restrict__ g_out,
                                 float* __restrict__ Wo) {
  const int row = blockIdx.x;                    // NS_*D_ rows, each length CD_
  const float* src = v_out + (size_t)row * CD_;
  float s = 0.f;
  for (int c = threadIdx.x; c < CD_; c += 32) { float x = src[c]; s += x * x; }
  s = warp_sum(s);
  float nrm = sqrtf(s); if (nrm < 1e-12f) nrm = 1e-12f;
  const float sc = g_out[row] / nrm;
  float* dst = Wo + (size_t)row * CD_;
  for (int c = threadIdx.x; c < CD_; c += 32) dst[c] = src[c] * sc;
}

// l2-normalized codebook rows in f16 (feeds the f16 WMMA search matmul)
__global__ void prep_codebook(const float* __restrict__ cb, _Float16* __restrict__ cnh) {
  const int row = blockIdx.x;                    // NS_*CS_ rows, each length CD_
  const float* src = cb + (size_t)row * CD_;
  float s = 0.f;
  for (int c = threadIdx.x; c < CD_; c += 32) { float x = src[c]; s += x * x; }
  s = warp_sum(s);
  float nrm = sqrtf(s); if (nrm < 1e-12f) nrm = 1e-12f;
  const float sc = 1.0f / nrm;
  _Float16* dst = cnh + (size_t)row * CD_;
  for (int c = threadIdx.x; c < CD_; c += 32) dst[c] = (_Float16)(src[c] * sc);
}

__global__ void zero_init(float* __restrict__ zq, float* __restrict__ scalars) {
  const size_t n = (size_t)B_ * D_ * T_;
  for (size_t i = (size_t)blockIdx.x * blockDim.x + threadIdx.x; i < n;
       i += (size_t)gridDim.x * blockDim.x)
    zq[i] = 0.f;
  if (blockIdx.x == 0 && threadIdx.x == 0) { scalars[0] = 0.f; scalars[1] = 0.f; }
}

// ---------------------------------------------------------------- FFT band filter
__device__ __forceinline__ void fft4096_stages(float2* s, int tid) {
  #pragma unroll 1
  for (int stage = 0; stage < 12; ++stage) {
    const int half = 1 << stage;
    for (int k = tid; k < T_ / 2; k += 256) {
      const int grp = k >> stage;
      const int pos = k & (half - 1);
      const int i0 = (grp << (stage + 1)) + pos;
      const int i1 = i0 + half;
      const float ang = -3.14159265358979f * (float)pos / (float)half;
      float sw, cw; __sincosf(ang, &sw, &cw);
      const float2 a = s[i0], b = s[i1];
      const float2 w = make_float2(b.x * cw - b.y * sw, b.x * sw + b.y * cw);
      s[i0] = make_float2(a.x + w.x, a.y + w.y);
      s[i1] = make_float2(a.x - w.x, a.y - w.y);
    }
    __syncthreads();
  }
}

// One row (b,d) per workgroup: rFFT mask via full complex FFT with conjugate symmetry.
__global__ void __launch_bounds__(256) fft_bandfilter_kernel(const float* __restrict__ src,
                                                             float* __restrict__ dst, int keep) {
  __shared__ float2 s[T_];                        // 32 KB of 320 KB/WGP
  const int tid = threadIdx.x;
  const float* in = src + (size_t)blockIdx.x * T_;
  float* out = dst + (size_t)blockIdx.x * T_;
  for (int t = tid; t < T_; t += 256) {           // load with bit reversal (12-bit)
    const int r = (int)(__brev((unsigned)t) >> 20);
    s[r] = make_float2(in[t], 0.f);
  }
  __syncthreads();
  fft4096_stages(s, tid);
  // band mask (keep k<=fl and conjugate mirror) + conjugate (inverse = conj->fwd->conj)
  for (int t = tid; t < T_; t += 256) {
    float2 v = s[t];
    const bool keepbin = (t <= keep) || (t >= T_ - keep);
    v.x = keepbin ? v.x : 0.f;
    v.y = keepbin ? -v.y : 0.f;
    s[t] = v;
  }
  __syncthreads();
  for (int t = tid; t < T_; t += 256) {           // in-place bit-reverse permute
    const int r = (int)(__brev((unsigned)t) >> 20);
    if (r > t) { const float2 a = s[t]; const float2 b = s[r]; s[t] = b; s[r] = a; }
  }
  __syncthreads();
  fft4096_stages(s, tid);
  const float invN = 1.0f / (float)T_;
  for (int t = tid; t < T_; t += 256) out[t] = s[t].x * invN;
}

// ---------------------------------------------------------------- in-proj (f32 WMMA + TDM)
// z_e[b,c,t] = Wi[c,:] . zf[b,:,t] + b_in[c].  M=64, K=512, N=32 per block.
// K-panels staged into LDS by the Tensor Data Mover (wave 0 issues, s_wait_tensorcnt gates).
__global__ void __launch_bounds__(256) in_proj_wmma_kernel(
    const float* __restrict__ zf, const float* __restrict__ Wi,
    const float* __restrict__ bin, float* __restrict__ ze, _Float16* __restrict__ zeh) {
  __shared__ float sW[CD_][128];                  // 32 KB Wi K-panel (packed rows of 128)
  __shared__ float sZ[128][33];                   // 16.5 KB zf K-panel (TDM pads 1 DWORD/row)
  const int tid = threadIdx.x;
  const int wave = tid >> 5, lane = tid & 31;
  const int m = lane & 15, hi = lane >> 4;        // ISA 16x4 f32 A layout: lanes 0-15 K{0,1}, 16-31 K{2,3}
  const int mt = wave & 3, nt = wave >> 2;
  const int b = blockIdx.y;
  const int t0 = blockIdx.x * 32;
  const unsigned ldsW = (unsigned)(unsigned long long)(uintptr_t)&sW[0][0];
  const unsigned ldsZ = (unsigned)(unsigned long long)(uintptr_t)&sZ[0][0];
  v8f acc = {};
  for (int kp = 0; kp < D_; kp += 128) {
    if (wave == 0) {
      // Wi panel: 64 rows x 128 contiguous floats, row stride D_; packed in LDS.
      tdm_load_2d(ldsW, Wi + kp, /*tile_d0=*/128, /*tile_d1=*/CD_,
                  /*tensor_d0=*/D_, /*tensor_d1=*/CD_, /*stride0=*/D_,
                  /*pad_en=*/0, /*pad_interval=*/0);
      // zf panel: 128 rows x 32 contiguous floats, row stride T_;
      // pad 1 DWORD every 32 DWORDs -> LDS row pitch 33 (bank-conflict-free).
      tdm_load_2d(ldsZ, zf + ((size_t)b * D_ + kp) * T_ + t0, /*tile_d0=*/32, /*tile_d1=*/128,
                  /*tensor_d0=*/T_, /*tensor_d1=*/128, /*stride0=*/T_,
                  /*pad_en=*/1, /*pad_interval=*/4);
      __builtin_amdgcn_s_wait_tensorcnt(0);
    }
    __syncthreads();
    #pragma unroll
    for (int kk = 0; kk < 128; kk += 4) {
      const int ka = kk + 2 * hi;
      v2f a, bb;
      a.x = sW[mt * 16 + m][ka];     a.y = sW[mt * 16 + m][ka + 1];
      bb.x = sZ[ka][nt * 16 + m];    bb.y = sZ[ka + 1][nt * 16 + m];
      acc = __builtin_amdgcn_wmma_f32_16x16x4_f32(false, a, false, bb, (short)0, acc,
                                                  false, false);
    }
    __syncthreads();
  }
  #pragma unroll
  for (int r = 0; r < 8; ++r) {                   // C/D layout: VGPR r -> rows r, r+8
    const int row = mt * 16 + r + 8 * hi;
    const int col = t0 + nt * 16 + m;
    const float v = acc[r] + bin[row];
    ze[((size_t)b * CD_ + row) * T_ + col] = v;
    zeh[((size_t)b * T_ + col) * CD_ + row] = (_Float16)v;   // t-major f16 copy for search
  }
}

// ---------------------------------------------------------------- codebook search (f16 WMMA)
// scores[t, j] = z_e[:,t] . cn[j,:]  (argmax == argmax(-dist) of the reference).
// Each wave: one 16-row t tile, sweeps 64 N-tiles of 16 codes with v_wmma_f32_16x16x32_f16.
__global__ void __launch_bounds__(256) codebook_search_kernel(
    const _Float16* __restrict__ zeh, const _Float16* __restrict__ cnh,
    int* __restrict__ idxbuf, float* __restrict__ codes, int stage) {
  const int tid = threadIdx.x;
  const int wave = tid >> 5, lane = tid & 31;
  const int m = lane & 15, hi = lane >> 4;
  const int b = blockIdx.y;
  const int t0 = (blockIdx.x * 8 + wave) * 16;
  // A frags (16x32 f16 layout: lanes 0-15 hold K{0..7,16..23}, lanes 16-31 K{8..15,24..31})
  const _Float16* abase = zeh + ((size_t)b * T_ + t0 + m) * CD_;
  v16h a0, a1;
  #pragma unroll
  for (int q = 0; q < 8; ++q) {
    a0[q]     = abase[hi * 8 + q];
    a0[8 + q] = abase[16 + hi * 8 + q];
    a1[q]     = abase[32 + hi * 8 + q];
    a1[8 + q] = abase[48 + hi * 8 + q];
  }
  float best[8]; int bestj[8];
  #pragma unroll
  for (int r = 0; r < 8; ++r) { best[r] = -3.4e38f; bestj[r] = 0; }
  for (int j0 = 0; j0 < CS_; j0 += 16) {
    const _Float16* bbase = cnh + (size_t)(j0 + m) * CD_;
    v16h b0, b1;
    #pragma unroll
    for (int q = 0; q < 8; ++q) {
      b0[q]     = bbase[hi * 8 + q];
      b0[8 + q] = bbase[16 + hi * 8 + q];
      b1[q]     = bbase[32 + hi * 8 + q];
      b1[8 + q] = bbase[48 + hi * 8 + q];
    }
    v8f c = {};
    c = __builtin_amdgcn_wmma_f32_16x16x32_f16(false, a0, false, b0, (short)0, c, false, false);
    c = __builtin_amdgcn_wmma_f32_16x16x32_f16(false, a1, false, b1, (short)0, c, false, false);
    const int j = j0 + m;
    #pragma unroll
    for (int r = 0; r < 8; ++r)
      if (c[r] > best[r]) { best[r] = c[r]; bestj[r] = j; }
  }
  #pragma unroll
  for (int r = 0; r < 8; ++r) {                   // reduce across the 16 lanes of each half
    float v = best[r]; int j = bestj[r];
    #pragma unroll
    for (int off = 8; off; off >>= 1) {
      const float ov = __shfl_xor(v, off, 32);
      const int   oj = __shfl_xor(j, off, 32);
      if (ov > v || (ov == v && oj < j)) { v = ov; j = oj; }
    }
    if (m == 0) {
      const int t = t0 + r + 8 * hi;
      idxbuf[(size_t)b * T_ + t] = j;
      codes[((size_t)b * NS_ + stage) * T_ + t] = (float)j;
    }
  }
}

// ---------------------------------------------------------------- out-proj + losses (f32 WMMA)
// zq gather -> LDS, loss accumulation, z_q_i = Wo . zq + b_out, update z_q and residual.
__global__ void __launch_bounds__(256) out_proj_wmma_kernel(
    const float* __restrict__ Wo, const float* __restrict__ bout,
    const float* __restrict__ cb, const int* __restrict__ idxbuf,
    const float* __restrict__ ze, float* __restrict__ zq_out,
    float* __restrict__ residual, float* __restrict__ scalars, float inv_count) {
  __shared__ float sQ[CD_][33];
  const int tid = threadIdx.x;
  const int wave = tid >> 5, lane = tid & 31;
  const int m = lane & 15, hi = lane >> 4;
  const int b = blockIdx.y;
  const int t0 = blockIdx.x * 32;
  float lsum = 0.f;
  for (int x = tid; x < CD_ * 32; x += 256) {
    const int k = x >> 5, c = x & 31;
    const int t = t0 + c;
    const int j = idxbuf[(size_t)b * T_ + t];
    const float q = cb[(size_t)j * CD_ + k];
    sQ[k][c] = q;
    const float d = ze[((size_t)b * CD_ + k) * T_ + t] - q;
    lsum += d * d;
  }
  lsum = warp_sum(lsum);
  if (lane == 0) {                                // commitment and codebook losses share fwd value
    atomicAdd(&scalars[0], lsum * inv_count);
    atomicAdd(&scalars[1], lsum * inv_count);
  }
  __syncthreads();
  for (int it = 0; it < 8; ++it) {                // 64 tiles (32 M x 2 N), 8 per wave
    const int tile = wave * 8 + it;
    const int mt = tile >> 1, nt = tile & 1;
    const float* wrow = Wo + (size_t)(mt * 16 + m) * CD_;  // L2-resident, reused by all blocks
    v8f acc = {};
    #pragma unroll
    for (int kk = 0; kk < CD_; kk += 4) {
      const int ka = kk + 2 * hi;
      v2f a, bb;
      a.x = wrow[ka];                a.y = wrow[ka + 1];
      bb.x = sQ[ka][nt * 16 + m];    bb.y = sQ[ka + 1][nt * 16 + m];
      acc = __builtin_amdgcn_wmma_f32_16x16x4_f32(false, a, false, bb, (short)0, acc,
                                                  false, false);
    }
    #pragma unroll
    for (int r = 0; r < 8; ++r) {
      const int row = mt * 16 + r + 8 * hi;
      const int col = t0 + nt * 16 + m;
      const float v = acc[r] + bout[row];
      const size_t o = ((size_t)b * D_ + row) * T_ + col;
      zq_out[o] += v;                             // z_q accumulates across stages
      residual[o] -= v;                           // residual recurrence
    }
  }
}

// ---------------------------------------------------------------- host launch
extern "C" void kernel_launch(void* const* d_in, const int* in_sizes, int n_in,
                              void* d_out, int out_size, void* d_ws, size_t ws_size,
                              hipStream_t stream) {
  (void)in_sizes; (void)n_in; (void)out_size; (void)ws_size;
  const float* z        = (const float*)d_in[0];
  const float* v_in     = (const float*)d_in[1];
  const float* g_in     = (const float*)d_in[2];
  const float* b_in     = (const float*)d_in[3];
  const float* codebook = (const float*)d_in[4];
  const float* v_out    = (const float*)d_in[5];
  const float* g_out    = (const float*)d_in[6];
  const float* b_out    = (const float*)d_in[7];

  float* out     = (float*)d_out;
  float* zq_out  = out;                                   // [B, D, T]
  float* codes   = out + (size_t)B_ * D_ * T_;            // [B, NS, T] (indices as float)
  float* scalars = codes + (size_t)B_ * NS_ * T_;         // [commitment, codebook]

  char* w = (char*)d_ws;
  auto take = [&](size_t bytes) -> char* {
    char* p = w; w += (bytes + 255) & ~(size_t)255; return p;
  };
  float*    residual = (float*)take((size_t)B_ * D_ * T_ * 4);
  float*    zf       = (float*)take((size_t)B_ * D_ * T_ * 4);
  float*    ze       = (float*)take((size_t)B_ * CD_ * T_ * 4);
  _Float16* zeh      = (_Float16*)take((size_t)B_ * T_ * CD_ * 2);
  int*      idxbuf   = (int*)take((size_t)B_ * T_ * 4);
  float*    Wi_hat   = (float*)take((size_t)NS_ * CD_ * D_ * 4);
  float*    Wo_hat   = (float*)take((size_t)NS_ * D_ * CD_ * 4);
  _Float16* cnh      = (_Float16*)take((size_t)NS_ * CS_ * CD_ * 2);

  (void)hipMemcpyAsync(residual, z, (size_t)B_ * D_ * T_ * 4,
                       hipMemcpyDeviceToDevice, stream);
  zero_init<<<2048, 256, 0, stream>>>(zq_out, scalars);
  prep_in_weights<<<NS_ * CD_, 32, 0, stream>>>(v_in, g_in, Wi_hat);
  prep_out_weights<<<NS_ * D_, 32, 0, stream>>>(v_out, g_out, Wo_hat);
  prep_codebook<<<NS_ * CS_, 32, 0, stream>>>(codebook, cnh);

  const int scales[NS_] = {4, 2, 1, 1};
  const float inv_count = 1.0f / (float)((size_t)B_ * CD_ * T_);
  for (int i = 0; i < NS_; ++i) {
    const float* zsrc = residual;
    if (scales[i] > 1) {
      const int fl = (T_ / 2 + 1) / scales[i];    // rfft bins kept: 0..fl
      fft_bandfilter_kernel<<<B_ * D_, 256, 0, stream>>>(residual, zf, fl);
      zsrc = zf;
    }
    in_proj_wmma_kernel<<<dim3(T_ / 32, B_), 256, 0, stream>>>(
        zsrc, Wi_hat + (size_t)i * CD_ * D_, b_in + (size_t)i * CD_, ze, zeh);
    codebook_search_kernel<<<dim3(T_ / 128, B_), 256, 0, stream>>>(
        zeh, cnh + (size_t)i * CS_ * CD_, idxbuf, codes, i);
    out_proj_wmma_kernel<<<dim3(T_ / 32, B_), 256, 0, stream>>>(
        Wo_hat + (size_t)i * D_ * CD_, b_out + (size_t)i * D_,
        codebook + (size_t)i * CS_ * CD_, idxbuf, ze, zq_out, residual,
        scalars, inv_count);
  }
}
